// GCN_78580721648121
// MI455X (gfx1250) — compile-verified, matrix operand order checked by reading, not run on hardware
//
#include <hip/hip_runtime.h>

// ---------------------------------------------------------------------------
// GCN 3-layer forward for MI455X (gfx1250, wave32, WMMA).
//   support = h @ W          -> v_wmma_f32_16x16x32_bf16 (bf16 in, f32 acc)
//   agg[d] += w_e*support[s] -> global_atomic_add_f32 (L2-resident dest)
//   h' = relu(agg + b) -> bf16 ; final: log_softmax(agg + b3)
// GEMM: ping-pong double-buffered K pipeline (no rotation movs), 32x64 strip
// per wave, compile-time K=512 so loads use immediate offsets.
// ---------------------------------------------------------------------------

typedef __attribute__((ext_vector_type(16))) __bf16 v16bf;
typedef __attribute__((ext_vector_type(8)))  __bf16 v8bf;
typedef __attribute__((ext_vector_type(8)))  float  v8f;

#define N_NODES 20000
#define N_EDGES 320000
#define NFEAT   512
#define NHID    512
#define NCLASS  64

__device__ __forceinline__ v16bf frag_cat(v8bf lo, v8bf hi) {
  return __builtin_shufflevector(lo, hi, 0, 1, 2, 3, 4, 5, 6, 7,
                                 8, 9, 10, 11, 12, 13, 14, 15);
}

// ---------------- conversion / utility kernels ----------------

__global__ void k_f32_to_bf16(const float* __restrict__ in,
                              __bf16* __restrict__ out, int n) {
  int i = blockIdx.x * blockDim.x + threadIdx.x;
  if (i < n) out[i] = (__bf16)in[i];
}

// W: K x N row-major (fp32)  ->  Wt: N x K row-major (bf16)
__global__ void k_transpose_to_bf16(const float* __restrict__ W,
                                    __bf16* __restrict__ Wt, int K, int N) {
  int i = blockIdx.x * blockDim.x + threadIdx.x;
  if (i < K * N) {
    int n = i / K;
    int k = i - n * K;
    Wt[i] = (__bf16)W[k * N + n];
  }
}

__global__ void k_zero_f32x4(float* __restrict__ p, int n4) {
  int i = blockIdx.x * blockDim.x + threadIdx.x;
  if (i < n4) ((float4*)p)[i] = make_float4(0.f, 0.f, 0.f, 0.f);
}

// ---------------- WMMA GEMM: C[M,N] = A[M,K] * Bt[N,K]^T ----------------
// A, Bt row-major bf16, K compile-time (512). Each wave: 32x64 strip
// (2 M-tiles x 4 N-tiles, 8 f32 accumulators); B fragments shared across
// both M-tiles. Ping-pong buffers: loads for step k+32 issue before the
// WMMAs of step k; buffer swap is static renaming (no v_mov rotation).
// Fragment layout (ISA 7.12.2, 16-bit A 16x32): lane&15 = row, lane>>4 picks
// the K octet; elems 0-7 = K[kb..kb+7], elems 8-15 = K[16+kb..16+kb+7].

template <int K>
__global__ void __launch_bounds__(256)
k_gemm_bf16_wmma(const __bf16* __restrict__ A, const __bf16* __restrict__ Bt,
                 float* __restrict__ C, int N, int totalWaves) {
  int wave = blockIdx.x * (blockDim.x >> 5) + (threadIdx.x >> 5);
  if (wave >= totalWaves) return;
  int tilesN = N >> 6;                  // 64-wide strips
  int tm = wave / tilesN;               // 32-row strip index
  int tn = wave - tm * tilesN;

  int lane = threadIdx.x & 31;
  int r    = lane & 15;                 // row (A) / col (B) within tile
  int kb   = (lane >> 4) << 3;          // 0 or 8: K-octet select

  const __bf16* a0 = A + (size_t)(tm * 32 + r) * K + kb;        // M-tile 0
  const __bf16* a1 = a0 + (size_t)16 * K;                       // M-tile 1
  const __bf16* b  = Bt + (size_t)(tn * 64 + r) * K + kb;

  v8f acc[2][4] = {};

  // [buf][mtile] / [buf][ntile] staged fragments
  v8bf Alo[2][2], Ahi[2][2], Blo[2][4], Bhi[2][4];

  auto loadFrags = [&](int buf, int k) {
    Alo[buf][0] = *(const v8bf*)(a0 + k);
    Ahi[buf][0] = *(const v8bf*)(a0 + k + 16);
    Alo[buf][1] = *(const v8bf*)(a1 + k);
    Ahi[buf][1] = *(const v8bf*)(a1 + k + 16);
#pragma unroll
    for (int t = 0; t < 4; ++t) {
      const __bf16* bp = b + (size_t)(t * 16) * K + k;
      Blo[buf][t] = *(const v8bf*)(bp);
      Bhi[buf][t] = *(const v8bf*)(bp + 16);
    }
  };

  auto compute = [&](int buf) {
    v16bf av0 = frag_cat(Alo[buf][0], Ahi[buf][0]);
    v16bf av1 = frag_cat(Alo[buf][1], Ahi[buf][1]);
#pragma unroll
    for (int t = 0; t < 4; ++t) {
      v16bf bv = frag_cat(Blo[buf][t], Bhi[buf][t]);
      acc[0][t] = __builtin_amdgcn_wmma_f32_16x16x32_bf16(
          false, av0, false, bv, (short)0, acc[0][t], false, false);
      acc[1][t] = __builtin_amdgcn_wmma_f32_16x16x32_bf16(
          false, av1, false, bv, (short)0, acc[1][t], false, false);
    }
  };

  loadFrags(0, 0);                      // prologue: stage k = 0

#pragma unroll 1
  for (int k = 0; k < K; k += 64) {
    loadFrags(1, k + 32);               // stage odd slice (k+32 < K always)
    __builtin_prefetch(a0 + k + 64, 0, 1);   // global_prefetch_b8
    __builtin_prefetch(a1 + k + 64, 0, 1);
    compute(0);                         // consume even slice
    int kn = (k + 64 < K) ? (k + 64) : k;    // tail: hot dummy reload
    loadFrags(0, kn);                   // stage next even slice
    compute(1);                         // consume odd slice
  }

  // C/D layout: VGPR v, lanes 0-15 -> row v, lanes 16-31 -> row v+8; col=lane&15
  int rowOff = (lane >> 4) << 3;
  int col    = lane & 15;
#pragma unroll
  for (int mt = 0; mt < 2; ++mt) {
#pragma unroll
    for (int t = 0; t < 4; ++t) {
#pragma unroll
      for (int v = 0; v < 8; ++v) {
        C[(size_t)(tm * 32 + mt * 16 + rowOff + v) * N + tn * 64 + t * 16 + col]
            = acc[mt][t][v];
      }
    }
  }
}

// ---------------- sparse aggregation: agg[dst] += w_e * support[src] -------
// one block per edge; float4 gathers, 4 f32 atomics per 16B load

__global__ void k_scatter(const float* __restrict__ sup,
                          const int* __restrict__ src,
                          const int* __restrict__ dst,
                          const float* __restrict__ w,
                          float* __restrict__ agg, int F) {
  int e = blockIdx.x;
  int s = src[e];
  int d = dst[e];
  float we = w[e];
  const float* sp = sup + (size_t)s * F;
  float*       ap = agg + (size_t)d * F;
  for (int f = threadIdx.x * 4; f < F; f += blockDim.x * 4) {
    float4 v = *(const float4*)(sp + f);
    atomicAdd(ap + f + 0, we * v.x);
    atomicAdd(ap + f + 1, we * v.y);
    atomicAdd(ap + f + 2, we * v.z);
    atomicAdd(ap + f + 3, we * v.w);
  }
}

// ---------------- epilogues ----------------

__global__ void k_bias_relu_bf16(const float* __restrict__ agg,
                                 const float* __restrict__ b,
                                 __bf16* __restrict__ out, int F, int n) {
  int i = blockIdx.x * blockDim.x + threadIdx.x;
  if (i < n) {
    float v = agg[i] + b[i & (F - 1)];   // F is a power of two (512)
    out[i] = (__bf16)(v > 0.f ? v : 0.f);
  }
}

// one wave per node, 64 classes -> 2 per lane; wave32 shuffle reductions
__global__ void k_logsoftmax64(const float* __restrict__ agg,
                               const float* __restrict__ b,
                               float* __restrict__ out, int M) {
  int node = blockIdx.x * (blockDim.x >> 5) + (threadIdx.x >> 5);
  int lane = threadIdx.x & 31;
  if (node >= M) return;
  const float* row = agg + (size_t)node * NCLASS;
  float v0 = row[lane]      + b[lane];
  float v1 = row[lane + 32] + b[lane + 32];
  float mx = fmaxf(v0, v1);
#pragma unroll
  for (int off = 16; off > 0; off >>= 1)
    mx = fmaxf(mx, __shfl_xor(mx, off, 32));
  float s = __expf(v0 - mx) + __expf(v1 - mx);
#pragma unroll
  for (int off = 16; off > 0; off >>= 1)
    s += __shfl_xor(s, off, 32);
  float lse = mx + __logf(s);
  out[(size_t)node * NCLASS + lane]      = v0 - lse;
  out[(size_t)node * NCLASS + lane + 32] = v1 - lse;
}

// ---------------- orchestration ----------------

extern "C" void kernel_launch(void* const* d_in, const int* in_sizes, int n_in,
                              void* d_out, int out_size, void* d_ws, size_t ws_size,
                              hipStream_t stream) {
  const float* x    = (const float*)d_in[0];
  const int*   esrc = (const int*)  d_in[1];
  const int*   edst = (const int*)  d_in[2];
  const float* ew   = (const float*)d_in[3];
  const float* W1   = (const float*)d_in[4];
  const float* b1   = (const float*)d_in[5];
  const float* W2   = (const float*)d_in[6];
  const float* b2   = (const float*)d_in[7];
  const float* W3   = (const float*)d_in[8];
  const float* b3   = (const float*)d_in[9];

  char* ws = (char*)d_ws;
  // 256B-aligned slices
  __bf16* h16 = (__bf16*)(ws);                                  // 20,480,000 B
  __bf16* w16 = (__bf16*)(ws + 20480000);                       //    524,288 B
  float*  sup = (float*) (ws + 20480000 + 524288);              // 40,960,000 B
  float*  agg = (float*) (ws + 20480000 + 524288 + 40960000);   // 40,960,000 B

  // x -> bf16
  {
    int n = N_NODES * NFEAT;
    k_f32_to_bf16<<<(n + 255) / 256, 256, 0, stream>>>(x, h16, n);
  }

  auto layer = [&](const float* W, const float* b, int Nout, bool reluOut) {
    // W (512 x Nout) -> Wt bf16 (Nout x 512)
    int wn = NHID * Nout;
    k_transpose_to_bf16<<<(wn + 255) / 256, 256, 0, stream>>>(W, w16, NHID, Nout);

    // GEMM: sup = h16 * Wt^T  (32x64 strip per wave, K=512 compile-time)
    int totalWaves = (N_NODES / 32) * (Nout / 64);
    k_gemm_bf16_wmma<NHID><<<(totalWaves + 7) / 8, 256, 0, stream>>>(
        h16, w16, sup, Nout, totalWaves);

    // zero + scatter
    int n4 = (N_NODES * Nout) / 4;
    k_zero_f32x4<<<(n4 + 255) / 256, 256, 0, stream>>>(agg, n4);
    int bs = Nout / 4;                  // 128 for F=512
    if (bs < 32) bs = 32;
    if (bs > 256) bs = 256;
    k_scatter<<<N_EDGES, bs, 0, stream>>>(sup, esrc, edst, ew, agg, Nout);

    if (reluOut) {
      int n = N_NODES * Nout;
      k_bias_relu_bf16<<<(n + 255) / 256, 256, 0, stream>>>(agg, b, h16, Nout, n);
    }
  };

  layer(W1, b1, NHID, true);
  layer(W2, b2, NHID, true);
  layer(W3, b3, NCLASS, false);

  // logits = agg + b3 ; out = log_softmax(logits)
  k_logsoftmax64<<<(N_NODES + 7) / 8, 256, 0, stream>>>(agg, b3, (float*)d_out, N_NODES);
}
